// BandSplitModule_77876347011702
// MI455X (gfx1250) — matrix operand order repaired and checked by reading, not compiled, not run on hardware
//
#include <hip/hip_runtime.h>

#define NBANDS 41
#define T_DIM  517
#define NFREQ  1025
#define BATCH  32
#define FC     128
#define LN_EPS 1e-5f
#define MT_PER_BLK 11   // 3 blocks * 11 tiles * 16 rows = 528 >= 517

// STFT-bin boundaries of the 41 bands (from freq2bands(BANDSPLITS, 44100, 2048))
__constant__ int d_bnd[NBANDS + 1] = {
    0, 4, 9, 13, 18, 23, 27, 32, 37, 41, 46,
    58, 69, 81, 92, 104, 116, 127, 139, 150, 162, 174, 185,
    208, 232, 255, 278, 301, 325, 348, 371,
    417, 464, 510, 557, 603, 650, 696, 742,
    835, 928, 1025};

struct BandPtrs {
    const float* lnw[NBANDS];
    const float* lnb[NBANDS];
    const float* fcw[NBANDS];
    const float* fcb[NBANDS];
};

typedef __attribute__((ext_vector_type(16))) __bf16 v16bf;
typedef __attribute__((ext_vector_type(8)))  __bf16 v8bf;
typedef __attribute__((ext_vector_type(8)))  float  v8f;

// ---------------- Kernel 1: per-(batch, band) mean / rstd ----------------
__global__ __launch_bounds__(256) void band_stats(const float* __restrict__ x,
                                                  float* __restrict__ stats) {
    const int b    = blockIdx.x;
    const int band = blockIdx.y;
    const int s    = d_bnd[band];
    const int w    = d_bnd[band + 1] - s;
    const int n    = w * T_DIM;                       // contiguous region
    const float* p = x + b * (NFREQ * T_DIM) + s * T_DIM;

    float sum = 0.f, sq = 0.f;
    for (int i = threadIdx.x; i < n; i += 256) {
        float v = p[i];
        sum += v;
        sq  += v * v;
    }
    __shared__ float s1[256], s2[256];
    const int tid = threadIdx.x;
    s1[tid] = sum;
    s2[tid] = sq;
    __syncthreads();
    for (int off = 128; off > 0; off >>= 1) {
        if (tid < off) { s1[tid] += s1[tid + off]; s2[tid] += s2[tid + off]; }
        __syncthreads();
    }
    if (tid == 0) {
        float inv = 1.f / (float)n;
        float mu  = s1[0] * inv;
        float var = s2[0] * inv - mu * mu;
        stats[(band * BATCH + b) * 2 + 0] = mu;
        stats[(band * BATCH + b) * 2 + 1] = rsqrtf(var + LN_EPS);
    }
}

// ---------------- Kernel 2: normalize + einsum via bf16 WMMA ----------------
__global__ __launch_bounds__(256) void band_gemm(const float* __restrict__ x,
                                                 const float* __restrict__ stats,
                                                 BandPtrs P,
                                                 float* __restrict__ out) {
    const int band = blockIdx.y;
    const int b    = blockIdx.z;
    const int s    = d_bnd[band];
    const int w    = d_bnd[band + 1] - s;
    const int nkt  = (w + 31) >> 5;       // K-tiles of 32
    const int kpad = nkt << 5;            // 32..128
    const int bstr = kpad + 8;            // LDS row stride (bf16 elems), breaks bank alignment

    __shared__ __bf16 As[16 * (128 + 8)];
    __shared__ __bf16 Bs[FC * (128 + 8)];

    const int tid  = threadIdx.x;
    const int lane = tid & 31;
    const int wv   = tid >> 5;      // wave 0..7 -> f tile
    const int nn   = lane & 15;
    const int half = lane >> 4;

    const float mu   = stats[(band * BATCH + b) * 2 + 0];
    const float rstd = stats[(band * BATCH + b) * 2 + 1];

    const float* fcw = P.fcw[band];
    const float* lnw = P.lnw[band];
    const float* lnb = P.lnb[band];
    const float* fcb = P.fcb[band];

    // Stage B = fc_w[f][k] (bf16, zero-padded to kpad); one wave per row, lanes stride k.
    for (int f = wv; f < FC; f += 8)
        for (int k = lane; k < kpad; k += 32) {
            float v = (k < w) ? fcw[f * w + k] : 0.f;
            Bs[f * bstr + k] = (__bf16)v;
        }

    const int   fbase = wv * 16;
    const float bias  = fcb[fbase + nn];
    const float* xb   = x + b * (NFREQ * T_DIM) + s * T_DIM;

    for (int mt = 0; mt < MT_PER_BLK; ++mt) {
        const int t0 = (blockIdx.x * MT_PER_BLK + mt) * 16;
        if (t0 >= T_DIM) break;                 // uniform over block
        __syncthreads();                        // As/Bs readers of prev iter done

        // Stage A = normalized xn, LDS layout [t][k].
        // tid>>4 strides k (coalesced global reads along t), tid&15 = t row.
        {
            const int  m   = tid & 15;
            const int  t   = t0 + m;
            const bool tin = (t < T_DIM);
            for (int k = tid >> 4; k < kpad; k += 16) {
                float v = 0.f;
                if (tin && k < w) {
                    int idx = k * T_DIM + t;
                    v = (xb[idx] - mu) * rstd * lnw[idx] + lnb[idx];
                }
                As[m * bstr + k] = (__bf16)v;
            }
        }
        __syncthreads();

        v8f acc = {};
        for (int kt = 0; kt < nkt; ++kt) {
            // A fragment (16x32 bf16): lane m=nn, element j -> K = kt*32 + half*8 + (j>>3)*16 + (j&7)
            const __bf16* ap = &As[nn * bstr + kt * 32 + half * 8];
            v8bf alo = *(const v8bf*)(ap);        // K offsets 0..7
            v8bf ahi = *(const v8bf*)(ap + 16);   // K offsets 16..23
            v16bf a = __builtin_shufflevector(alo, ahi,
                0, 1, 2, 3, 4, 5, 6, 7, 8, 9, 10, 11, 12, 13, 14, 15);

            // B fragment (32x16 bf16): lane n=nn(f), element j -> K = kt*32 + half*8 + (j>>3)*16 + (j&7)
            const __bf16* bp = &Bs[(fbase + nn) * bstr + kt * 32 + half * 8];
            v8bf blo = *(const v8bf*)(bp);
            v8bf bhi = *(const v8bf*)(bp + 16);
            v16bf bf = __builtin_shufflevector(blo, bhi,
                0, 1, 2, 3, 4, 5, 6, 7, 8, 9, 10, 11, 12, 13, 14, 15);

            acc = __builtin_amdgcn_wmma_f32_16x16x32_bf16(
                false, a, false, bf, (short)0, acc, false, false);
        }

        // D layout: VGPR v -> row m = v + half*8, col f = fbase + nn
        float* op = out + ((b * NBANDS + band) * T_DIM + t0) * FC + fbase + nn;
#pragma unroll
        for (int v = 0; v < 8; ++v) {
            int m = v + half * 8;
            if (t0 + m < T_DIM) op[m * FC] = acc[v] + bias;
        }
    }
}

extern "C" void kernel_launch(void* const* d_in, const int* in_sizes, int n_in,
                              void* d_out, int out_size, void* d_ws, size_t ws_size,
                              hipStream_t stream) {
    const float* x = (const float*)d_in[0];
    BandPtrs P;
    for (int i = 0; i < NBANDS; ++i) {
        P.lnw[i] = (const float*)d_in[1 + i];
        P.lnb[i] = (const float*)d_in[1 + NBANDS + i];
        P.fcw[i] = (const float*)d_in[1 + 2 * NBANDS + i];
        P.fcb[i] = (const float*)d_in[1 + 3 * NBANDS + i];
    }
    float* stats = (float*)d_ws;   // 41*32*2 floats = 10.5 KB

    band_stats<<<dim3(BATCH, NBANDS), 256, 0, stream>>>(x, stats);
    band_gemm <<<dim3(3, NBANDS, BATCH), 256, 0, stream>>>(x, stats, P, (float*)d_out);
}
